// TrajectoryLSTM_64828236366574
// MI455X (gfx1250) — compile-verified, matrix operand order checked by reading, not run on hardware
//
#include <hip/hip_runtime.h>
#include <hip/hip_bf16.h>

// ---------------------------------------------------------------------------
// Seq2seq LSTM + attention for MI455X (gfx1250), wave32 + WMMA f16->f32.
// All matmuls run on v_wmma_f32_16x16x32_f16 with operands pre-packed into
// exact fragment lane/slot order (CDNA5 ISA 7.12.2), so inner loops are
// pure b128-load -> wmma.
// ---------------------------------------------------------------------------

typedef __attribute__((ext_vector_type(16))) _Float16 v16h;
typedef __attribute__((ext_vector_type(8)))  float    v8f;

#define Hh   256
#define Bb   512
#define Ss   128
#define INN  10
#define OUTT 3
#define TT   35
#define MIDD 128

// ---- fragment index helpers (A: rows m x 32 K; B mirrored with n) ----------
__device__ __forceinline__ int frag_lane(int mn, int k) {
  return mn + (((k & 15) >= 8) ? 16 : 0);
}
__device__ __forceinline__ int frag_slot(int k) {
  int kk = k & 7;
  return 2 * ((kk >> 1) + 4 * (k >> 4)) + (kk & 1);
}

__device__ __forceinline__ v8f wmma_acc(const _Float16* ap, const _Float16* bp,
                                        int KT, v8f c, int lane) {
  for (int k = 0; k < KT; ++k) {
    v16h a = *(const v16h*)(ap + (k * 32 + lane) * 16);
    v16h b = *(const v16h*)(bp + (k * 32 + lane) * 16);
    c = __builtin_amdgcn_wmma_f32_16x16x32_f16(false, a, false, b, (short)0, c,
                                               false, false);
  }
  return c;
}

__device__ __forceinline__ float sigmf(float x) { return 1.f / (1.f + expf(-x)); }

// ---------------------------------------------------------------------------
__global__ void zero_u32(unsigned int* p, long n) {
  long i = (long)blockIdx.x * blockDim.x + threadIdx.x;
  if (i < n) p[i] = 0u;
}

// Pack f32 weight W (rows = output cols N, row stride ldw, col offset coloff)
// into WMMA B-fragment layout [ntile][ktile][lane][slot].
// mode 0: src col = gk if gk < Ksrc else 0
// mode 1 (cWih): gk<3 -> gk ; 32<=gk<288 -> gk-29 ; else 0   (din pad to 32)
__global__ void pack_weight(const float* W, int ldw, int coloff, int Ksrc,
                            int KT, _Float16* dst, int mode, long total) {
  long tid = (long)blockIdx.x * blockDim.x + threadIdx.x;
  if (tid >= total) return;
  int r2 = (int)(tid & 511);
  long tile = tid >> 9;
  int ktile = (int)(tile % KT);
  long ntile = tile / KT;
  int n = r2 & 15, k = r2 >> 4;
  int gk = ktile * 32 + k, srck;
  if (mode == 1) srck = (gk < 3) ? gk : ((gk >= 32) ? gk - 29 : -1);
  else           srck = (gk < Ksrc) ? gk : -1;
  float v = (srck >= 0) ? W[(ntile * 16 + n) * (long)ldw + coloff + srck] : 0.f;
  dst[tile * 512 + frag_lane(n, k) * 16 + frag_slot(k)] = (_Float16)v;
}

// Pack x (B,S,IN) -> A-fragments [s][mtile][lane][slot], K padded 10->32.
__global__ void pack_x(const float* x, _Float16* dst) {
  long tid = (long)blockIdx.x * blockDim.x + threadIdx.x;
  if (tid >= 128L * 32 * 512) return;
  int r2 = (int)(tid & 511);
  long tile = tid >> 9;
  int mtile = (int)(tile % 32);
  int s = (int)(tile / 32);
  int m = r2 & 15, k = r2 >> 4;
  int b = mtile * 16 + m;
  float v = (k < INN) ? x[((long)b * Ss + s) * INN + k] : 0.f;
  dst[tile * 512 + frag_lane(m, k) * 16 + frag_slot(k)] = (_Float16)v;
}

// Pack f32 (512 x 256) row-major -> A-fragments, KT=8.
__global__ void pack_f32_rows(const float* src, _Float16* dst) {
  int tid = blockIdx.x * blockDim.x + threadIdx.x;  // 131072
  int b = tid >> 8, k = tid & 255;
  int mtile = b >> 4, m = b & 15, ktile = k >> 5, kk = k & 31;
  dst[((long)(mtile * 8 + ktile)) * 512 + frag_lane(m, kk) * 16 + frag_slot(kk)] =
      (_Float16)src[tid];
}

// ---------------------------------------------------------------------------
// Fused LSTM step: grid (32 mtiles, 16 jtiles), block 128 (4 waves = i,f,g,o).
// gates = Xfrag@Wx + Hfrag@Wh + b0 + b1 ; cell update ; write c, h(f32/pack).
// ---------------------------------------------------------------------------
__global__ void lstm_step(const _Float16* xap, const _Float16* xbp, int kx,
                          const _Float16* hap, const _Float16* hbp, int kh,
                          const float* b0, const float* b1,
                          const float* c_in, float* c_out,
                          _Float16* hpack0, _Float16* hpack1,
                          float* hf32, long hstride) {
  __shared__ float g[4][256];
  int mtile = blockIdx.x, jtile = blockIdx.y;
  int wave = threadIdx.x >> 5, lane = threadIdx.x & 31;
  int ntile_g = wave * 16 + jtile;  // gate col tile in [0,64)
  int gcol = ntile_g * 16 + (lane & 15);
  float bias = b0[gcol] + b1[gcol];
  v8f c;
  for (int r = 0; r < 8; ++r) c[r] = bias;
  if (xap) c = wmma_acc(xap + (long)mtile * kx * 512, xbp + (long)ntile_g * kx * 512, kx, c, lane);
  c = wmma_acc(hap + (long)mtile * kh * 512, hbp + (long)ntile_g * kh * 512, kh, c, lane);
  int hi = lane >> 4, n0 = lane & 15;
  for (int r = 0; r < 8; ++r) g[wave][(r + 8 * hi) * 16 + n0] = c[r];
  __syncthreads();
  for (int e = threadIdx.x; e < 256; e += 128) {
    int m = e >> 4, n = e & 15;
    int b = mtile * 16 + m, j = jtile * 16 + n;
    float iv = g[0][e], fv = g[1][e], gv = g[2][e], ov = g[3][e];
    float cold = c_in[(long)b * Hh + j];
    float cn = sigmf(fv) * cold + sigmf(iv) * tanhf(gv);
    float h = sigmf(ov) * tanhf(cn);
    c_out[(long)b * Hh + j] = cn;
    int ktile = j >> 5, kk = j & 31;
    long pidx = ((long)(mtile * 8 + ktile)) * 512 + frag_lane(m, kk) * 16 + frag_slot(kk);
    if (hpack0) hpack0[pidx] = (_Float16)h;
    if (hpack1) hpack1[pidx] = (_Float16)h;
    if (hf32) hf32[(long)b * hstride + j] = h;
  }
}

// ---------------------------------------------------------------------------
// Generic WMMA GEMM + bias + activation. grid (32, Ntiles, Z), block 32.
// act: 0 none, 1 tanh, 2 gelu(exact). Optional f32 out and A-frag pack out.
// ---------------------------------------------------------------------------
__global__ void gemm_act(const _Float16* ap, long astride, int KT,
                         const _Float16* bp, const float* bias, int act,
                         float* outf, long zoff, long rowstride,
                         _Float16* packout, int packKT) {
  int mtile = blockIdx.x, ntile = blockIdx.y, z = blockIdx.z;
  int lane = threadIdx.x;
  float bv = bias ? bias[ntile * 16 + (lane & 15)] : 0.f;
  v8f c;
  for (int r = 0; r < 8; ++r) c[r] = bv;
  c = wmma_acc(ap + (long)z * astride + (long)mtile * KT * 512,
               bp + (long)ntile * KT * 512, KT, c, lane);
  int n = lane & 15, hi = lane >> 4;
  for (int r = 0; r < 8; ++r) {
    float v = c[r];
    if (act == 1) v = tanhf(v);
    else if (act == 2) v = 0.5f * v * (1.f + erff(v * 0.70710678118f));
    int row = mtile * 16 + r + 8 * hi;
    int col = ntile * 16 + n;
    if (outf) outf[(long)z * zoff + (long)row * rowstride + col] = v;
    if (packout) {
      int ktile = col >> 5, kk = col & 31;
      packout[((long)(mtile * packKT + ktile)) * 512 + frag_lane(row & 15, kk) * 16 +
              frag_slot(kk)] = (_Float16)v;
    }
  }
}

// ---------------------------------------------------------------------------
// Attention: per-b workgroup (256 thr). energy/softmax/context; pack ctx into
// xin fragments (k=32+d) and oi fragments (k=256+d).
// ---------------------------------------------------------------------------
__global__ void attn_kernel(const float* ahid, const float* encproj,
                            const float* encf, const float* vW,
                            _Float16* xinpack, _Float16* oipack) {
  int b = blockIdx.x;
  int tid = threadIdx.x, wave = tid >> 5, lane = tid & 31;
  __shared__ float eng[128], wsm[128], stats[2];
  for (int s = wave; s < 128; s += 8) {
    float p = 0.f;
    for (int i = 0; i < 8; ++i) {
      int d = lane + 32 * i;
      p += tanhf(encproj[((long)b * Ss + s) * Hh + d] + ahid[(long)b * Hh + d]) * vW[d];
    }
    for (int off = 16; off > 0; off >>= 1) p += __shfl_down(p, off, 32);
    if (lane == 0) eng[s] = p;
  }
  __syncthreads();
  if (tid == 0) {
    float mx = eng[0];
    for (int s = 1; s < 128; ++s) mx = fmaxf(mx, eng[s]);
    float sm = 0.f;
    for (int s = 0; s < 128; ++s) sm += expf(eng[s] - mx);
    stats[0] = mx; stats[1] = sm;
  }
  __syncthreads();
  if (tid < 128) wsm[tid] = expf(eng[tid] - stats[0]) / stats[1];
  __syncthreads();
  int d = tid;  // 0..255
  float acc = 0.f;
  for (int s = 0; s < 128; ++s) acc += wsm[s] * encf[((long)b * Ss + s) * Hh + d];
  int mtile = b >> 4, m = b & 15, kk = d & 31;
  _Float16 hv = (_Float16)acc;
  xinpack[((long)(mtile * 9 + 1 + (d >> 5))) * 512 + frag_lane(m, kk) * 16 + frag_slot(kk)] = hv;
  oipack[((long)(mtile * 16 + 8 + (d >> 5))) * 512 + frag_lane(m, kk) * 16 + frag_slot(kk)] = hv;
}

// ---------------------------------------------------------------------------
// LayerNorm over H=256 per row; writes decoder-h fragments and oi fragments.
// ---------------------------------------------------------------------------
__global__ void dec_ln(const float* h2raw, const float* lng, const float* lnb,
                       _Float16* hpack, _Float16* oipack) {
  int b = blockIdx.x, d = threadIdx.x;
  float x = h2raw[(long)b * Hh + d];
  int wave = d >> 5, lane = d & 31;
  __shared__ float s1[8], s2[8], mv[2];
  float a = x, q = x * x;
  for (int off = 16; off > 0; off >>= 1) {
    a += __shfl_down(a, off, 32);
    q += __shfl_down(q, off, 32);
  }
  if (lane == 0) { s1[wave] = a; s2[wave] = q; }
  __syncthreads();
  if (d == 0) {
    float A = 0.f, Q = 0.f;
    for (int i = 0; i < 8; ++i) { A += s1[i]; Q += s2[i]; }
    float mu = A / 256.f;
    mv[0] = mu; mv[1] = rsqrtf(Q / 256.f - mu * mu + 1e-5f);
  }
  __syncthreads();
  float h = (x - mv[0]) * mv[1] * lng[d] + lnb[d];
  int mtile = b >> 4, m = b & 15, ktile = d >> 5, kk = d & 31;
  _Float16 hv = (_Float16)h;
  hpack [((long)(mtile * 8  + ktile)) * 512 + frag_lane(m, kk) * 16 + frag_slot(kk)] = hv;
  oipack[((long)(mtile * 16 + ktile)) * 512 + frag_lane(m, kk) * 16 + frag_slot(kk)] = hv;
}

// Output head: pred = mid @ p2W^T + p2b ; writes d_out[b,t,:] and din frags.
__global__ void p2_kernel(const float* mid, const float* p2W, const float* p2b,
                          float* out, int t, _Float16* xinpack) {
  int idx = blockIdx.x * blockDim.x + threadIdx.x;
  if (idx >= Bb * OUTT) return;
  int b = idx / OUTT, o = idx % OUTT;
  float acc = p2b[o];
  for (int i = 0; i < MIDD; ++i) acc += mid[(long)b * MIDD + i] * p2W[o * MIDD + i];
  out[((long)b * TT + t) * OUTT + o] = acc;
  xinpack[((long)((b >> 4) * 9)) * 512 + frag_lane(b & 15, o) * 16 + frag_slot(o)] =
      (_Float16)acc;
}

// ---------------------------------------------------------------------------
extern "C" void kernel_launch(void* const* d_in, const int* in_sizes, int n_in,
                              void* d_out, int out_size, void* d_ws, size_t ws_size,
                              hipStream_t stream) {
  (void)in_sizes; (void)n_in; (void)out_size; (void)ws_size;
  const float* x     = (const float*)d_in[0];
  const float* eWih0 = (const float*)d_in[1];
  const float* eWhh0 = (const float*)d_in[2];
  const float* ebih0 = (const float*)d_in[3];
  const float* ebhh0 = (const float*)d_in[4];
  const float* eWih1 = (const float*)d_in[5];
  const float* eWhh1 = (const float*)d_in[6];
  const float* ebih1 = (const float*)d_in[7];
  const float* ebhh1 = (const float*)d_in[8];
  const float* aW    = (const float*)d_in[9];
  const float* ab    = (const float*)d_in[10];
  const float* vW    = (const float*)d_in[11];
  const float* cWih  = (const float*)d_in[12];
  const float* cWhh  = (const float*)d_in[13];
  const float* cbih  = (const float*)d_in[14];
  const float* cbhh  = (const float*)d_in[15];
  const float* lng   = (const float*)d_in[16];
  const float* lnb   = (const float*)d_in[17];
  const float* p1W   = (const float*)d_in[18];
  const float* p1b   = (const float*)d_in[19];
  const float* p2W   = (const float*)d_in[20];
  const float* p2b   = (const float*)d_in[21];
  const float* hW    = (const float*)d_in[22];
  const float* hb    = (const float*)d_in[23];
  const float* cW2   = (const float*)d_in[24];
  const float* cb2   = (const float*)d_in[25];
  float* out = (float*)d_out;

  char* p = (char*)d_ws;
  auto alloc = [&](size_t bytes) -> char* {
    char* r = p; p += (bytes + 255) / 256 * 256; return r;
  };
  // f16 fragment buffers
  _Float16* Wih0p   = (_Float16*)alloc(64L * 1 * 512 * 2);
  _Float16* Whh0p   = (_Float16*)alloc(64L * 8 * 512 * 2);
  _Float16* Wih1p   = (_Float16*)alloc(64L * 8 * 512 * 2);
  _Float16* Whh1p   = (_Float16*)alloc(64L * 8 * 512 * 2);
  _Float16* aWencp  = (_Float16*)alloc(16L * 8 * 512 * 2);
  _Float16* aWhidp  = (_Float16*)alloc(16L * 8 * 512 * 2);
  _Float16* cWihp   = (_Float16*)alloc(64L * 9 * 512 * 2);
  _Float16* cWhhp   = (_Float16*)alloc(64L * 8 * 512 * 2);
  _Float16* p1Wp    = (_Float16*)alloc(8L * 16 * 512 * 2);
  _Float16* hWp     = (_Float16*)alloc(16L * 8 * 512 * 2);
  _Float16* cW2p    = (_Float16*)alloc(16L * 8 * 512 * 2);
  _Float16* xpack   = (_Float16*)alloc(128L * 32 * 512 * 2);
  _Float16* h0A     = (_Float16*)alloc(131072L * 2);
  _Float16* h0B     = (_Float16*)alloc(131072L * 2);
  _Float16* h1A     = (_Float16*)alloc(131072L * 2);
  _Float16* h1B     = (_Float16*)alloc(131072L * 2);
  _Float16* enc1pk  = (_Float16*)alloc(128L * 131072 * 2);
  _Float16* encpk   = (_Float16*)alloc(128L * 131072 * 2);
  _Float16* dechpk  = (_Float16*)alloc(131072L * 2);
  _Float16* cnpack  = (_Float16*)alloc(131072L * 2);
  _Float16* xinpack = (_Float16*)alloc(32L * 9 * 512 * 2);
  _Float16* oipack  = (_Float16*)alloc(32L * 16 * 512 * 2);
  // f32 buffers
  float* c0      = (float*)alloc(131072L * 4);
  float* c1      = (float*)alloc(131072L * 4);
  float* decc    = (float*)alloc(131072L * 4);
  float* encf    = (float*)alloc(512L * 128 * 256 * 4);
  float* encproj = (float*)alloc(512L * 128 * 256 * 4);
  float* ahid    = (float*)alloc(131072L * 4);
  float* h2raw   = (float*)alloc(131072L * 4);
  float* mid     = (float*)alloc(512L * 128 * 4);

  auto zero = [&](void* ptr, long u32s) {
    zero_u32<<<(unsigned)((u32s + 255) / 256), 256, 0, stream>>>((unsigned int*)ptr, u32s);
  };
  zero(h0A, 65536); zero(h1A, 65536);
  zero(c0, 131072); zero(c1, 131072);
  zero(xinpack, 32L * 9 * 512 / 2);

  auto pw = [&](const float* W, int ldw, int coloff, int Ksrc, int Ntiles, int KT,
                _Float16* dst, int mode) {
    long total = (long)Ntiles * KT * 512;
    pack_weight<<<(unsigned)((total + 255) / 256), 256, 0, stream>>>(
        W, ldw, coloff, Ksrc, KT, dst, mode, total);
  };
  pw(eWih0, 10, 0, 10, 64, 1, Wih0p, 0);
  pw(eWhh0, 256, 0, 256, 64, 8, Whh0p, 0);
  pw(eWih1, 256, 0, 256, 64, 8, Wih1p, 0);
  pw(eWhh1, 256, 0, 256, 64, 8, Whh1p, 0);
  pw(aW, 512, 0, 256, 16, 8, aWencp, 0);
  pw(aW, 512, 256, 256, 16, 8, aWhidp, 0);
  pw(cWih, 259, 0, 259, 64, 9, cWihp, 1);
  pw(cWhh, 256, 0, 256, 64, 8, cWhhp, 0);
  pw(p1W, 512, 0, 512, 8, 16, p1Wp, 0);
  pw(hW, 256, 0, 256, 16, 8, hWp, 0);
  pw(cW2, 256, 0, 256, 16, 8, cW2p, 0);
  pack_x<<<8192, 256, 0, stream>>>(x, xpack);

  dim3 lgrid(32, 16);
  // ---- encoder layer 0 ----
  {
    _Float16* hin = h0A; _Float16* hout = h0B;
    for (int t = 0; t < Ss; ++t) {
      lstm_step<<<lgrid, 128, 0, stream>>>(
          xpack + (long)t * 32 * 512, Wih0p, 1, hin, Whh0p, 8,
          ebih0, ebhh0, c0, c0, hout, enc1pk + (long)t * 131072, nullptr, 0);
      _Float16* tmp = hin; hin = hout; hout = tmp;
    }
  }
  // ---- encoder layer 1 (also writes enc f32 + enc fragments) ----
  _Float16* h1fin;
  {
    _Float16* hin = h1A; _Float16* hout = h1B;
    for (int t = 0; t < Ss; ++t) {
      lstm_step<<<lgrid, 128, 0, stream>>>(
          enc1pk + (long)t * 131072, Wih1p, 8, hin, Whh1p, 8,
          ebih1, ebhh1, c1, c1, hout, encpk + (long)t * 131072,
          encf + (long)t * Hh, (long)Ss * Hh);
      _Float16* tmp = hin; hin = hout; hout = tmp;
    }
    h1fin = hin;  // last written
  }
  // ---- enc_proj = enc @ aW_enc^T + ab  (all timesteps in one launch) ----
  gemm_act<<<dim3(32, 16, 128), 32, 0, stream>>>(
      encpk, 131072L, 8, aWencp, ab, 0, encproj, 256L, (long)Ss * Hh, nullptr, 0);
  // ---- decoder init: h0 = tanh(h_n@hW^T+hb) -> frags; c0 = c_n@cW2^T+cb2 ----
  gemm_act<<<dim3(32, 16, 1), 32, 0, stream>>>(
      h1fin, 0L, 8, hWp, hb, 1, nullptr, 0L, 0L, dechpk, 8);
  pack_f32_rows<<<512, 256, 0, stream>>>(c1, cnpack);
  gemm_act<<<dim3(32, 16, 1), 32, 0, stream>>>(
      cnpack, 0L, 8, cW2p, cb2, 0, decc, 0L, 256L, nullptr, 0);
  // ---- decoder loop ----
  for (int t = 0; t < TT; ++t) {
    gemm_act<<<dim3(32, 16, 1), 32, 0, stream>>>(           // ahid = h @ aW_hid^T
        dechpk, 0L, 8, aWhidp, nullptr, 0, ahid, 0L, 256L, nullptr, 0);
    attn_kernel<<<512, 256, 0, stream>>>(ahid, encproj, encf, vW, xinpack, oipack);
    lstm_step<<<lgrid, 128, 0, stream>>>(                   // decoder cell
        xinpack, cWihp, 9, dechpk, cWhhp, 8, cbih, cbhh,
        decc, decc, nullptr, nullptr, h2raw, 256L);
    dec_ln<<<512, 256, 0, stream>>>(h2raw, lng, lnb, dechpk, oipack);
    gemm_act<<<dim3(32, 8, 1), 32, 0, stream>>>(            // mid = gelu(oi@p1W^T+b)
        oipack, 0L, 16, p1Wp, p1b, 2, mid, 0L, 128L, nullptr, 0);
    p2_kernel<<<6, 256, 0, stream>>>(mid, p2W, p2b, out, t, xinpack);
  }
}
